// SelfAttention_48928267436576
// MI455X (gfx1250) — matrix operand order repaired and checked by reading, not compiled, not run on hardware
//
#include <hip/hip_runtime.h>

typedef __attribute__((ext_vector_type(16))) __bf16 v16bf;
typedef __attribute__((ext_vector_type(8)))  __bf16 v8bf;
typedef __attribute__((ext_vector_type(4)))  __bf16 v4bf;
typedef __attribute__((ext_vector_type(8)))  float  v8f;
typedef __attribute__((ext_vector_type(4)))  unsigned int v4u;
typedef __attribute__((ext_vector_type(8)))  int v8i;
typedef __attribute__((ext_vector_type(4)))  int v4i;

namespace {
constexpr int kB = 2, kS = 2048, kH = 16, kD = 64;
constexpr int QTILE  = 128;            // q rows per workgroup
constexpr int KVTILE = 32;             // kv rows per staged tile
constexpr int NWAVES = QTILE / 16;     // 8 waves (wave32)
constexpr float kScale = 0.125f;       // 1/sqrt(64), folded into Q pre-pass
constexpr float kNegInf = -3.0e38f;
}

#if defined(__has_builtin)
#if __has_builtin(__builtin_amdgcn_tensor_load_to_lds) && __has_builtin(__builtin_amdgcn_s_wait_tensorcnt)
#define HAVE_TDM 1
#endif
#endif

// ---------------- DPP16 butterfly reduction within 16-lane groups ----------------
// xor1 = quad_perm(1,0,3,2)=0xB1, xor2 = quad_perm(2,3,0,1)=0x4E,
// then row_half_mirror (0x141) and row_mirror (0x140) complete the 16-lane reduce.
template <int CTRL>
__device__ __forceinline__ float dpp_max(float x) {
  int y = __builtin_amdgcn_update_dpp(__float_as_int(x), __float_as_int(x),
                                      CTRL, 0xf, 0xf, true);
  return fmaxf(x, __int_as_float(y));
}
template <int CTRL>
__device__ __forceinline__ float dpp_add(float x) {
  int y = __builtin_amdgcn_update_dpp(__float_as_int(x), __float_as_int(x),
                                      CTRL, 0xf, 0xf, true);
  return x + __int_as_float(y);
}
__device__ __forceinline__ float red16_max(float x) {
  x = dpp_max<0xB1>(x); x = dpp_max<0x4E>(x);
  x = dpp_max<0x141>(x); x = dpp_max<0x140>(x);
  return x;
}
__device__ __forceinline__ float red16_add(float x) {
  x = dpp_add<0xB1>(x); x = dpp_add<0x4E>(x);
  x = dpp_add<0x141>(x); x = dpp_add<0x140>(x);
  return x;
}

#ifdef HAVE_TDM
// 2D tensor-tile DMA: global (bf16 elements) -> LDS, via Tensor Data Mover.
__device__ __forceinline__ void tdm_load_2d(unsigned lds_addr, const __bf16* gptr,
                                            unsigned tensor_d0, unsigned tensor_d1,
                                            unsigned stride0,
                                            unsigned tile_d0, unsigned tile_d1) {
  const unsigned long long ga = (unsigned long long)(uintptr_t)gptr;
  v4u g0;
  g0[0] = 1u;                                          // count=1, user mode
  g0[1] = lds_addr;                                    // LDS byte address
  g0[2] = (unsigned)ga;                                // global_addr[31:0]
  g0[3] = ((unsigned)(ga >> 32) & 0x01FFFFFFu)         // global_addr[56:32]
          | 0x80000000u;                               // type=2 ("image")
  v8i g1;
  g1[0] = 0x00010000;                                  // data_size=1 (2B), no mask
  g1[1] = (int)(tensor_d0 << 16);                      // tensor_dim0[15:0]
  g1[2] = (int)((tensor_d0 >> 16) | (tensor_d1 << 16));// dim0[31:16] | dim1[15:0]
  g1[3] = (int)((tensor_d1 >> 16) | (tile_d0 << 16));  // dim1[31:16] | tile_dim0
  g1[4] = (int)(tile_d1 & 0xFFFFu);                    // tile_dim1, tile_dim2=0
  g1[5] = (int)stride0;                                // tensor_dim0_stride[31:0]
  g1[6] = 0;
  g1[7] = 0;
  const v4i gz4 = {0, 0, 0, 0};
  const v8i gz8 = {0, 0, 0, 0, 0, 0, 0, 0};
  __builtin_amdgcn_tensor_load_to_lds(g0, g1, gz4, gz4, gz8, 0);
}
#endif

// ---- pre-pass: fp32 qkv -> bf16 Q*scale[B,H,S,D], K[B,H,S,D], V^T[B,H,D,S] ----
__global__ __launch_bounds__(256)
void cvt_kernel(const float* __restrict__ qkv, __bf16* __restrict__ qbf,
                __bf16* __restrict__ kbf, __bf16* __restrict__ vbft) {
  const int i  = blockIdx.x * 256 + threadIdx.x;   // over [B,S,H,D/4]
  const int d4 = (i & 15) * 4;
  const int h  = (i >> 4) & 15;
  const int s  = (i >> 8) & (kS - 1);
  const int b  = i >> 19;
  const float4 qv = *(const float4*)(qkv + ((((b * kS + s) * 3 + 0) * kH + h) * kD + d4));
  const float4 kv = *(const float4*)(qkv + ((((b * kS + s) * 3 + 1) * kH + h) * kD + d4));
  const float4 vv = *(const float4*)(qkv + ((((b * kS + s) * 3 + 2) * kH + h) * kD + d4));
  const int bh = b * kH + h;
  const v4bf q4 = { (__bf16)(qv.x * kScale), (__bf16)(qv.y * kScale),
                    (__bf16)(qv.z * kScale), (__bf16)(qv.w * kScale) };
  const v4bf k4 = { (__bf16)kv.x, (__bf16)kv.y, (__bf16)kv.z, (__bf16)kv.w };
  *(v4bf*)(qbf + (bh * kS + s) * kD + d4) = q4;
  *(v4bf*)(kbf + (bh * kS + s) * kD + d4) = k4;
  __bf16* vp = vbft + (bh * kD + d4) * kS + s;     // transposed scatter
  vp[0 * kS] = (__bf16)vv.x;  vp[1 * kS] = (__bf16)vv.y;
  vp[2 * kS] = (__bf16)vv.z;  vp[3 * kS] = (__bf16)vv.w;
}

// ---------------- flash-attention forward ----------------
__global__ __launch_bounds__(NWAVES * 32)
void fa_fwd_kernel(const __bf16* __restrict__ qbf, const __bf16* __restrict__ kbf,
                   const __bf16* __restrict__ vbft, float* __restrict__ out) {
  __shared__ __align__(32) __bf16 KtBuf[2][KVTILE][kD];        // [kv][d], 2x4 KB
  __shared__ __align__(32) __bf16 VtBuf[2][kD][KVTILE];        // [d][kv], 2x4 KB
  __shared__ __align__(32) __bf16 Pt[NWAVES][16][KVTILE];      // 8 KB

  const int tid  = threadIdx.x;
  const int wave = tid >> 5;
  const int lane = tid & 31;
  const int lh   = lane >> 4;
  const int ln   = lane & 15;

  const int bh    = blockIdx.y;          // b*H + h
  const int qbase = blockIdx.x * QTILE;
  const int qw    = qbase + wave * 16;

  const __bf16* kslice = kbf  + bh * (kS * kD);
  const __bf16* vslice = vbft + bh * (kD * kS);

  // ---- Q fragments (A-matrix 16x32 bf16 layout; scale pre-folded) ----
  const int qrow = qw + ln;
  const __bf16* qp = qbf + (bh * kS + qrow) * kD;
  v16bf qa[2];
  #pragma unroll
  for (int s = 0; s < 2; ++s) {
    const int d0 = 32 * s + lh * 16;
    const v8bf lo = *(const v8bf*)(qp + d0);
    const v8bf hi = *(const v8bf*)(qp + d0 + 8);
    #pragma unroll
    for (int j = 0; j < 8; ++j) { qa[s][j] = lo[j]; qa[s][8 + j] = hi[j]; }
  }

  // ---- online-softmax state ----
  float rowmax[8], rowsum[8];
  v8f acc[4];
  const v8f vzero = {};
  #pragma unroll
  for (int r = 0; r < 8; ++r) { rowmax[r] = kNegInf; rowsum[r] = 0.0f; }
  #pragma unroll
  for (int nt = 0; nt < 4; ++nt) acc[nt] = vzero;

  const int ntiles = (qbase + QTILE) / KVTILE;

#ifdef HAVE_TDM
  if (wave == 0) {   // prefetch tile 0 (EXEC ignored by TDM; one issue per wave)
    tdm_load_2d((unsigned)(uintptr_t)&KtBuf[0][0][0], kslice, kD, kS, kD, kD, KVTILE);
    tdm_load_2d((unsigned)(uintptr_t)&VtBuf[0][0][0], vslice, kS, kD, kS, KVTILE, kD);
  }
#endif

  for (int t = 0; t < ntiles; ++t) {
    const int kv0 = t * KVTILE;
    const int buf = t & 1;

#ifdef HAVE_TDM
    if (wave == 0) {
      if (t + 1 < ntiles) {       // overlap: issue DMA for tile t+1, then wait tile t
        const int nb = (t + 1) & 1;
        const int nkv = (t + 1) * KVTILE;
        tdm_load_2d((unsigned)(uintptr_t)&KtBuf[nb][0][0], kslice + nkv * kD,
                    kD, kS, kD, kD, KVTILE);
        tdm_load_2d((unsigned)(uintptr_t)&VtBuf[nb][0][0], vslice + nkv,
                    kS, kD, kS, KVTILE, kD);
        __builtin_amdgcn_s_wait_tensorcnt(2);
      } else {
        __builtin_amdgcn_s_wait_tensorcnt(0);
      }
    }
    __syncthreads();
#else
    __syncthreads();
    {  // cooperative bf16 staging: 256 threads x 16B each per tile
      const int row = tid >> 3, col = (tid & 7) * 8;         // K: 32x64
      *(v8bf*)&KtBuf[buf][row][col] =
          *(const v8bf*)(kslice + (kv0 + row) * kD + col);
      const int d = tid >> 2, c = (tid & 3) * 8;             // V^T: 64x32
      *(v8bf*)&VtBuf[buf][d][c] =
          *(const v8bf*)(vslice + d * kS + kv0 + c);
    }
    __syncthreads();
#endif

    if (kv0 < qw + 16) {   // not fully masked for this wave
      // ---- load all four K B-fragments, then run the WMMA chain ----
      v16bf kb[2][2];
      #pragma unroll
      for (int c = 0; c < 2; ++c)
        #pragma unroll
        for (int s = 0; s < 2; ++s)
          kb[c][s] = *(const v16bf*)&KtBuf[buf][c * 16 + ln][32 * s + lh * 16];

      v8f sc[2];
      #pragma unroll
      for (int c = 0; c < 2; ++c) {
        v8f cc = {};
        #pragma unroll
        for (int s = 0; s < 2; ++s)
          cc = __builtin_amdgcn_wmma_f32_16x16x32_bf16(
              false, qa[s], false, kb[c][s], (short)0, cc, false, false);
        sc[c] = cc;
      }

      // ---- causal mask only on diagonal tiles (wave-uniform branch) ----
      if (kv0 + KVTILE - 1 > qw) {
        #pragma unroll
        for (int r = 0; r < 8; ++r) {
          const int qi = qw + r + 8 * lh;
          if (kv0 + ln > qi)      sc[0][r] = kNegInf;
          if (kv0 + 16 + ln > qi) sc[1][r] = kNegInf;
        }
      }

      // ---- row max (DPP butterfly, no LDS traffic) ----
      float nval[8];
      #pragma unroll
      for (int r = 0; r < 8; ++r)
        nval[r] = red16_max(fmaxf(sc[0][r], sc[1][r]));

      // ---- exponentiate, rescale accumulators, spill P to LDS ----
      float psum[8];
      #pragma unroll
      for (int r = 0; r < 8; ++r) {
        const float mnew = fmaxf(rowmax[r], nval[r]);
        const float corr = __expf(rowmax[r] - mnew);
        rowmax[r] = mnew;
        const float p0 = __expf(sc[0][r] - mnew);
        const float p1 = __expf(sc[1][r] - mnew);
        psum[r] = p0 + p1;
        rowsum[r] *= corr;
        #pragma unroll
        for (int nt = 0; nt < 4; ++nt) acc[nt][r] *= corr;
        const int row = r + 8 * lh;
        Pt[wave][row][ln]      = (__bf16)p0;
        Pt[wave][row][16 + ln] = (__bf16)p1;
      }
      #pragma unroll
      for (int r = 0; r < 8; ++r) rowsum[r] += red16_add(psum[r]);

      // ---- P x V: load P and all four V B-fragments, then WMMA chain ----
      const v8bf plo = *(const v8bf*)&Pt[wave][ln][lh * 16];
      const v8bf phi = *(const v8bf*)&Pt[wave][ln][lh * 16 + 8];
      v16bf pa;
      #pragma unroll
      for (int j = 0; j < 8; ++j) { pa[j] = plo[j]; pa[8 + j] = phi[j]; }

      v16bf vb[4];
      #pragma unroll
      for (int nt = 0; nt < 4; ++nt)
        vb[nt] = *(const v16bf*)&VtBuf[buf][nt * 16 + ln][lh * 16];
      #pragma unroll
      for (int nt = 0; nt < 4; ++nt)
        acc[nt] = __builtin_amdgcn_wmma_f32_16x16x32_bf16(
            false, pa, false, vb[nt], (short)0, acc[nt], false, false);
    }
    __syncthreads();   // all waves done with buf before it is overwritten
  }

  // ---- epilogue: normalize, store fp32 [B,S,H,D] ----
  const int b = bh >> 4, h = bh & 15;
  #pragma unroll
  for (int r = 0; r < 8; ++r) {
    const float inv = 1.0f / rowsum[r];
    const int qi = qw + r + 8 * lh;
    float* op = out + (((b * kS + qi) * kH + h) * kD);
    #pragma unroll
    for (int nt = 0; nt < 4; ++nt)
      op[nt * 16 + ln] = acc[nt][r] * inv;
  }
}

extern "C" void kernel_launch(void* const* d_in, const int* in_sizes, int n_in,
                              void* d_out, int out_size, void* d_ws, size_t ws_size,
                              hipStream_t stream) {
  (void)in_sizes; (void)n_in; (void)out_size; (void)ws_size;
  const float* qkv = (const float*)d_in[0];
  float* out = (float*)d_out;

  const int per = kB * kH * kS * kD;               // 4,194,304 elements
  __bf16* qbf  = (__bf16*)d_ws;                    // 8 MB
  __bf16* kbf  = qbf + per;                        // 8 MB
  __bf16* vbft = kbf + per;                        // 8 MB (transposed V)

  cvt_kernel<<<(kB * kS * kH * (kD / 4)) / 256, 256, 0, stream>>>(qkv, qbf, kbf, vbft);

  dim3 grid(kS / QTILE, kB * kH);                  // (16, 32)
  fa_fwd_kernel<<<grid, NWAVES * 32, 0, stream>>>(qbf, kbf, vbft, out);
}
